// TopkRouting_1700807049483
// MI455X (gfx1250) — matrix operand order repaired and checked by reading, not compiled, not run on hardware
//
#include <hip/hip_runtime.h>
#include <math.h>

// ---------------------------------------------------------------------------
// TopkRouting fused kernel for gfx1250 (MI455X, wave32, WMMA).
//   logits = (q * 128^-0.5) @ k^T   (per batch, 1024x1024, bf16 WMMA, f32 acc)
//   per-row top-16 (value desc, index asc tie-break) + softmax over the 16.
// One workgroup (256 thr = 8 waves) owns one (batch, 16-query-row) strip.
// Logits never touch HBM: 16 x 1024 f32 strip lives in LDS (bank-padded).
// ---------------------------------------------------------------------------

typedef __attribute__((ext_vector_type(16))) __bf16 bf16x16;
typedef __attribute__((ext_vector_type(8)))  float  f32x8;
typedef __attribute__((ext_vector_type(4)))  float  f32x4;

#define QK_DIM  128
#define NP2     1024
#define TOPK    16
#define LSTRIDE 1028                    // 1024 + 4 dwords: de-conflict LDS banks
#define SCALE   0.08838834764831845f    // 128^-0.5

__global__ __launch_bounds__(256)
void topk_routing_kernel(const float* __restrict__ q,
                         const float* __restrict__ kmat,
                         float* __restrict__ out_w,
                         float* __restrict__ out_i)
{
    extern __shared__ float lds[];      // [16][LSTRIDE] logits strip

    const int tid   = threadIdx.x;
    const int lane  = tid & 31;
    const int wid   = tid >> 5;         // wave 0..7
    const int n     = blockIdx.x >> 6;  // batch index
    const int qt    = blockIdx.x & 63;  // 16-row query tile within batch
    const int qBase = qt * 16;

    // ---------------- Phase 1: WMMA logits into LDS -----------------------
    const int m    = lane & 15;         // row / column index within 16-tile
    const int half = lane >> 4;         // 0: lanes 0-15, 1: lanes 16-31

    // A fragments: 16x128 q-rows, scaled, f32 -> bf16.
    // ISA layout (16-bit A 16x32): lane<16 -> M=lane, K = {0..7, 16..23};
    //                              lane>=16 -> M=lane-16, K = {8..15, 24..31}.
    const float* qrow = q + ((size_t)n * NP2 + qBase + m) * QK_DIM;
    bf16x16 afrag[4];
#pragma unroll
    for (int f = 0; f < 4; ++f) {
        const int k0 = f * 32 + half * 8;
        const f32x4 lo0 = *(const f32x4*)(qrow + k0);
        const f32x4 lo1 = *(const f32x4*)(qrow + k0 + 4);
        const f32x4 hi0 = *(const f32x4*)(qrow + k0 + 16);
        const f32x4 hi1 = *(const f32x4*)(qrow + k0 + 20);
#pragma unroll
        for (int e = 0; e < 4; ++e) {
            afrag[f][e]      = (__bf16)(lo0[e] * SCALE);
            afrag[f][4 + e]  = (__bf16)(lo1[e] * SCALE);
            afrag[f][8 + e]  = (__bf16)(hi0[e] * SCALE);
            afrag[f][12 + e] = (__bf16)(hi1[e] * SCALE);
        }
    }

    // B fragments: key rows are columns of B (B = k^T), so each lane reads a
    // contiguous run of one key row.  ISA layout (16-bit B 32x16):
    //   lanes 0-15: N=lane,  K={0..7,16..23} ; lanes 16-31: N=lane-16, K={8..15,24..31}.
    const float* kbase    = kmat + (size_t)n * NP2 * QK_DIM;
    const int    kOffLane = half * 8;

    for (int t = 0; t < 8; ++t) {
        const int    kt   = wid * 8 + t;                         // key tile
        const float* krow = kbase + (size_t)(kt * 16 + m) * QK_DIM;
        if (t < 7)
            __builtin_prefetch(krow + 16 * QK_DIM, 0, 0);        // next tile

        f32x8 acc = {};
#pragma unroll
        for (int f = 0; f < 4; ++f) {
            const int k0 = f * 32 + kOffLane;
            bf16x16 bfrag;
#pragma unroll
            for (int e4 = 0; e4 < 4; ++e4) {
                const f32x4 v = *(const f32x4*)(krow + k0 + (e4 >> 1) * 16 + (e4 & 1) * 4);
                bfrag[e4 * 4 + 0] = (__bf16)v[0];
                bfrag[e4 * 4 + 1] = (__bf16)v[1];
                bfrag[e4 * 4 + 2] = (__bf16)v[2];
                bfrag[e4 * 4 + 3] = (__bf16)v[3];
            }
            acc = __builtin_amdgcn_wmma_f32_16x16x32_bf16(
                      false, afrag[f], false, bfrag,
                      (short)0, acc, false, false);
        }

        // C layout: VGPR r -> row (half*8 + r), col = lane&15.
#pragma unroll
        for (int r = 0; r < 8; ++r)
            lds[(half * 8 + r) * LSTRIDE + kt * 16 + m] = acc[r];
    }

    __syncthreads();

    // ---------------- Phase 2: per-row top-16 + softmax -------------------
    // 16 groups of 16 lanes; group g owns query row g of the strip.
    const int g  = tid >> 4;
    const int tl = tid & 15;
    const float* row = lds + g * LSTRIDE;

    // Register-resident sorted (desc) top-16 of this lane's 64 strided cols.
    float vlist[TOPK];
    int   ilist[TOPK];
#pragma unroll
    for (int i = 0; i < TOPK; ++i) { vlist[i] = -INFINITY; ilist[i] = 0x7FFFFFFF; }

    for (int j = 0; j < NP2 / 16; ++j) {
        const int   idx = tl + 16 * j;
        const float v   = row[idx];
        if (v > vlist[TOPK - 1]) {              // fast reject
            float cv = v; int ci = idx;
#pragma unroll
            for (int p = 0; p < TOPK; ++p) {    // unrolled insertion bubble
                const bool sw = (cv > vlist[p]) || (cv == vlist[p] && ci < ilist[p]);
                const float tv = sw ? cv : vlist[p];
                const int   ti = sw ? ci : ilist[p];
                cv = sw ? vlist[p] : cv;
                ci = sw ? ilist[p] : ci;
                vlist[p] = tv; ilist[p] = ti;
            }
        }
    }

    // Merge 16 sorted lists by 16 rounds of cross-lane extract-max.
    // xor offsets 1,2,4,8 stay inside each 16-lane half of the wave32.
    float myV = 0.0f; int myIdx = 0;
    for (int i = 0; i < TOPK; ++i) {
        float bv = vlist[0];
        int   bi = ilist[0];
#pragma unroll
        for (int off = 8; off >= 1; off >>= 1) {
            const float ov = __shfl_xor(bv, off, 32);
            const int   oi = __shfl_xor(bi, off, 32);
            const bool take = (ov > bv) || (ov == bv && oi < bi);
            bv = take ? ov : bv;
            bi = take ? oi : bi;
        }
        if (tl == i) { myV = bv; myIdx = bi; }  // lane i keeps i-th winner
        if (ilist[0] == bi) {                   // winning lane pops its head
#pragma unroll
            for (int p = 0; p < TOPK - 1; ++p) { vlist[p] = vlist[p + 1]; ilist[p] = ilist[p + 1]; }
            vlist[TOPK - 1] = -INFINITY; ilist[TOPK - 1] = 0x7FFFFFFF;
        }
    }

    // Softmax over the 16 kept logits (group-lane 0 holds the max).
    const float topV = __shfl(myV, lane & 16, 32);
    float e = __expf(myV - topV);
    float s = e;
#pragma unroll
    for (int off = 8; off >= 1; off >>= 1)
        s += __shfl_xor(s, off, 32);
    const float w = e / s;

    const size_t base = (((size_t)n * NP2) + qBase + g) * TOPK + tl;
    out_w[base] = w;
    out_i[base] = (float)myIdx;     // 0..1023 exact in f32
}

extern "C" void kernel_launch(void* const* d_in, const int* in_sizes, int n_in,
                              void* d_out, int out_size, void* d_ws, size_t ws_size,
                              hipStream_t stream) {
    const float* q = (const float*)d_in[0];
    const float* k = (const float*)d_in[1];
    float* out = (float*)d_out;

    const int    N = in_sizes[0] / (NP2 * QK_DIM);   // batches (256)
    const size_t W = (size_t)N * NP2 * TOPK;         // r_weight element count

    const dim3   grid(N * (NP2 / 16));               // one block per 16-q strip
    const size_t shmem = (size_t)16 * LSTRIDE * sizeof(float);  // 64.25 KB

    topk_routing_kernel<<<grid, 256, shmem, stream>>>(q, k, out, out + W);
}
